// KMeans_59622736003644
// MI455X (gfx1250) — matrix operand order repaired and checked by reading, not compiled, not run on hardware
//
#include <hip/hip_runtime.h>

// ---------------------------------------------------------------------------
// KMeans assignment on MI455X (gfx1250, wave32, WMMA).
//
// labels[i] = argmin_k ( c2[k] - 2 * dot(X[i], C[k]) )      (x^2 term dropped:
// constant per row, argmin-invariant).
//
// GEMM done with V_WMMA_F32_16X16X32_BF16 using a bf16 hi/lo split of the
// fp32 inputs (3 WMMAs per fp32 tile => ~2^-16 relative error, fp32-grade
// labels).  Each wave owns a 32-row x 32-centroid register tile; X rows are
// read from HBM exactly once and staged (pre-split to bf16) in LDS.
// ---------------------------------------------------------------------------

typedef __attribute__((ext_vector_type(16))) __bf16 v16bf;
typedef __attribute__((ext_vector_type(8)))  float  v8f;

union ABf16 {
  v16bf v;
  uint4 u[2];
};

__device__ __forceinline__ unsigned int f32_to_bf16_rne(float f) {
  unsigned int u = __float_as_uint(f);
  return (u + 0x7FFFu + ((u >> 16) & 1u)) >> 16;   // round-to-nearest-even
}
__device__ __forceinline__ float bf16_bits_to_f32(unsigned int b) {
  return __uint_as_float(b << 16);
}

// float4 -> packed bf16 hi pair-dwords and bf16 lo pair-dwords
__device__ __forceinline__ void cvt4(const float4 v, uint2& ph, uint2& pl) {
  unsigned int h0 = f32_to_bf16_rne(v.x), h1 = f32_to_bf16_rne(v.y);
  unsigned int h2 = f32_to_bf16_rne(v.z), h3 = f32_to_bf16_rne(v.w);
  float l0 = v.x - bf16_bits_to_f32(h0), l1 = v.y - bf16_bits_to_f32(h1);
  float l2 = v.z - bf16_bits_to_f32(h2), l3 = v.w - bf16_bits_to_f32(h3);
  ph = make_uint2(h0 | (h1 << 16), h2 | (h3 << 16));
  pl = make_uint2(f32_to_bf16_rne(l0) | (f32_to_bf16_rne(l1) << 16),
                  f32_to_bf16_rne(l2) | (f32_to_bf16_rne(l3) << 16));
}

#define DIM   256
#define LDSTR 264   // padded LDS row stride (bf16 elems): rows hit distinct bank quads

// ---------------------------------------------------------------------------
// Prep: centroids fp32 -> bf16 hi/lo arrays + squared norms c2[k].
// One block per centroid row, 256 threads (one per dim).
// ---------------------------------------------------------------------------
__global__ __launch_bounds__(256) void kmeans_prep(
    const float* __restrict__ C,
    unsigned short* __restrict__ Chi,
    unsigned short* __restrict__ Clo,
    float* __restrict__ c2) {
  const int row = blockIdx.x;
  const int t   = threadIdx.x;
  float v = C[row * DIM + t];
  unsigned int hb = f32_to_bf16_rne(v);
  float lo = v - bf16_bits_to_f32(hb);
  unsigned int lb = f32_to_bf16_rne(lo);
  Chi[row * DIM + t] = (unsigned short)hb;
  Clo[row * DIM + t] = (unsigned short)lb;

  __shared__ float red[256];
  red[t] = v * v;
  __syncthreads();
#pragma unroll
  for (int ofs = 128; ofs > 0; ofs >>= 1) {
    if (t < ofs) red[t] += red[t + ofs];
    __syncthreads();
  }
  if (t == 0) c2[row] = red[0];
}

// ---------------------------------------------------------------------------
// Main: one wave (32 threads) per block; wave owns 32 X rows.
// ---------------------------------------------------------------------------
__global__ __launch_bounds__(32) void kmeans_assign(
    const float* __restrict__ X,
    const unsigned short* __restrict__ Chi,
    const unsigned short* __restrict__ Clo,
    const float* __restrict__ c2,
    int* __restrict__ out,
    int n, int kcent) {

  __shared__ __align__(16) unsigned short shh[32 * LDSTR];
  __shared__ __align__(16) unsigned short shl[32 * LDSTR];

  const int lane    = threadIdx.x;           // 0..31
  const int rowBase = blockIdx.x * 32;
  if (rowBase >= n) return;

  // ---- stage 32 rows of X into LDS as bf16 hi/lo (X read once from HBM) ----
  {
    const int cb = lane * 8;                 // this lane covers 8 dims per row
#pragma unroll 4
    for (int r = 0; r < 32; ++r) {
      int rr = rowBase + r; if (rr >= n) rr = n - 1;
      const float* src = X + (size_t)rr * DIM + cb;
      float4 v0 = *(const float4*)(src);
      float4 v1 = *(const float4*)(src + 4);
      uint2 ph0, pl0, ph1, pl1;
      cvt4(v0, ph0, pl0);
      cvt4(v1, ph1, pl1);
      unsigned short* dh = shh + r * LDSTR + cb;
      unsigned short* dl = shl + r * LDSTR + cb;
      *(uint4*)dh = make_uint4(ph0.x, ph0.y, ph1.x, ph1.y);  // ds_store_b128
      *(uint4*)dl = make_uint4(pl0.x, pl0.y, pl1.x, pl1.y);
    }
  }

  // A-matrix 16-bit 16x32 layout: lane -> row (lane&15); halves take
  // K offsets {0..7,16..23} (lanes 0-15) / {8..15,24..31} (lanes 16-31).
  const int nloc = lane & 15;
  const int hf   = lane >> 4;
  const unsigned short* a0h = shh + nloc        * LDSTR + hf * 8;
  const unsigned short* a0l = shl + nloc        * LDSTR + hf * 8;
  const unsigned short* a1h = shh + (16 + nloc) * LDSTR + hf * 8;
  const unsigned short* a1l = shl + (16 + nloc) * LDSTR + hf * 8;

  float mind0[8], mind1[8];
  int   mini0[8], mini1[8];
#pragma unroll
  for (int r = 0; r < 8; ++r) {
    mind0[r] = mind1[r] = 3.4e38f;
    mini0[r] = mini1[r] = 0;
  }

  const int nChunkPairs = kcent / 32;        // 32 centroids per outer iter
  for (int c = 0; c < nChunkPairs; ++c) {
    const int cent0 = c * 32 + nloc;         // B column owned by this lane
    const int cent1 = cent0 + 16;
    // B-matrix 16-bit 32x16 layout: lane -> column (lane&15); lanes 0-15
    // hold K {0..7,16..23}, lanes 16-31 hold K {8..15,24..31} (same as A).
    const unsigned short* b0h = Chi + (size_t)cent0 * DIM + hf * 8;
    const unsigned short* b0l = Clo + (size_t)cent0 * DIM + hf * 8;
    const unsigned short* b1h = Chi + (size_t)cent1 * DIM + hf * 8;
    const unsigned short* b1l = Clo + (size_t)cent1 * DIM + hf * 8;

    v8f acc00 = {}, acc01 = {}, acc10 = {}, acc11 = {};

#pragma unroll
    for (int s = 0; s < 8; ++s) {            // K = 256 in steps of 32
      const int kb = 32 * s;
      ABf16 ah0, al0, ah1, al1, bh, bl;
      ah0.u[0] = *(const uint4*)(a0h + kb);
      ah0.u[1] = *(const uint4*)(a0h + kb + 16);
      al0.u[0] = *(const uint4*)(a0l + kb);
      al0.u[1] = *(const uint4*)(a0l + kb + 16);
      ah1.u[0] = *(const uint4*)(a1h + kb);
      ah1.u[1] = *(const uint4*)(a1h + kb + 16);
      al1.u[0] = *(const uint4*)(a1l + kb);
      al1.u[1] = *(const uint4*)(a1l + kb + 16);

      // chunk 0 columns
      bh.u[0] = *(const uint4*)(b0h + kb);
      bh.u[1] = *(const uint4*)(b0h + kb + 16);
      bl.u[0] = *(const uint4*)(b0l + kb);
      bl.u[1] = *(const uint4*)(b0l + kb + 16);
      acc00 = __builtin_amdgcn_wmma_f32_16x16x32_bf16(false, ah0.v, false, bh.v, (short)0, acc00, false, false);
      acc10 = __builtin_amdgcn_wmma_f32_16x16x32_bf16(false, ah1.v, false, bh.v, (short)0, acc10, false, false);
      acc00 = __builtin_amdgcn_wmma_f32_16x16x32_bf16(false, ah0.v, false, bl.v, (short)0, acc00, false, false);
      acc10 = __builtin_amdgcn_wmma_f32_16x16x32_bf16(false, ah1.v, false, bl.v, (short)0, acc10, false, false);
      acc00 = __builtin_amdgcn_wmma_f32_16x16x32_bf16(false, al0.v, false, bh.v, (short)0, acc00, false, false);
      acc10 = __builtin_amdgcn_wmma_f32_16x16x32_bf16(false, al1.v, false, bh.v, (short)0, acc10, false, false);

      // chunk 1 columns
      bh.u[0] = *(const uint4*)(b1h + kb);
      bh.u[1] = *(const uint4*)(b1h + kb + 16);
      bl.u[0] = *(const uint4*)(b1l + kb);
      bl.u[1] = *(const uint4*)(b1l + kb + 16);
      acc01 = __builtin_amdgcn_wmma_f32_16x16x32_bf16(false, ah0.v, false, bh.v, (short)0, acc01, false, false);
      acc11 = __builtin_amdgcn_wmma_f32_16x16x32_bf16(false, ah1.v, false, bh.v, (short)0, acc11, false, false);
      acc01 = __builtin_amdgcn_wmma_f32_16x16x32_bf16(false, ah0.v, false, bl.v, (short)0, acc01, false, false);
      acc11 = __builtin_amdgcn_wmma_f32_16x16x32_bf16(false, ah1.v, false, bl.v, (short)0, acc11, false, false);
      acc01 = __builtin_amdgcn_wmma_f32_16x16x32_bf16(false, al0.v, false, bh.v, (short)0, acc01, false, false);
      acc11 = __builtin_amdgcn_wmma_f32_16x16x32_bf16(false, al1.v, false, bh.v, (short)0, acc11, false, false);
    }

    // d2 (minus x^2) = c2 - 2*dot ; running argmin.  cent0 < cent1 and cent
    // increases with c, so strict '<' keeps the first (smallest) index.
    const float c20 = c2[cent0];
    const float c21 = c2[cent1];
#pragma unroll
    for (int r = 0; r < 8; ++r) {
      float d;
      d = fmaf(-2.0f, acc00[r], c20); if (d < mind0[r]) { mind0[r] = d; mini0[r] = cent0; }
      d = fmaf(-2.0f, acc01[r], c21); if (d < mind0[r]) { mind0[r] = d; mini0[r] = cent1; }
      d = fmaf(-2.0f, acc10[r], c20); if (d < mind1[r]) { mind1[r] = d; mini1[r] = cent0; }
      d = fmaf(-2.0f, acc11[r], c21); if (d < mind1[r]) { mind1[r] = d; mini1[r] = cent1; }
    }
  }

  // Cross-lane argmin over the 16 columns spread across each 16-lane half
  // (xor masks < 16 stay within the half).  Ties -> smaller centroid index.
#pragma unroll
  for (int m = 1; m < 16; m <<= 1) {
#pragma unroll
    for (int r = 0; r < 8; ++r) {
      float od = __shfl_xor(mind0[r], m, 32);
      int   oi = __shfl_xor(mini0[r], m, 32);
      if (od < mind0[r] || (od == mind0[r] && oi < mini0[r])) { mind0[r] = od; mini0[r] = oi; }
      od = __shfl_xor(mind1[r], m, 32);
      oi = __shfl_xor(mini1[r], m, 32);
      if (od < mind1[r] || (od == mind1[r] && oi < mini1[r])) { mind1[r] = od; mini1[r] = oi; }
    }
  }

  // C/D layout: slot r <-> point rowBase + hf*8 + r (tile0) / +16 (tile1).
#pragma unroll
  for (int r = 0; r < 8; ++r) {
    if (nloc == r) {
      int p0 = rowBase + hf * 8 + r;
      int p1 = rowBase + 16 + hf * 8 + r;
      if (p0 < n) out[p0] = mini0[r];
      if (p1 < n) out[p1] = mini1[r];
    }
  }
}

// ---------------------------------------------------------------------------
// Launch: prep (tiny) then the WMMA assignment kernel.
// Workspace layout: Chi[K*256] u16 | Clo[K*256] u16 | c2[K] f32  (~1.05 MB).
// ---------------------------------------------------------------------------
extern "C" void kernel_launch(void* const* d_in, const int* in_sizes, int n_in,
                              void* d_out, int out_size, void* d_ws, size_t ws_size,
                              hipStream_t stream) {
  const float* X = (const float*)d_in[0];
  const float* C = (const float*)d_in[1];
  const int n = in_sizes[0] / DIM;       // 200000
  const int k = in_sizes[1] / DIM;       // 1024

  unsigned short* Chi = (unsigned short*)d_ws;
  unsigned short* Clo = Chi + (size_t)k * DIM;
  float*          c2  = (float*)(Clo + (size_t)k * DIM);
  int*            out = (int*)d_out;     // argmin labels (int32)

  kmeans_prep<<<k, 256, 0, stream>>>(C, Chi, Clo, c2);

  const int nGroups = (n + 31) / 32;     // 6250 waves, 32 rows each
  kmeans_assign<<<nGroups, 32, 0, stream>>>(X, Chi, Clo, c2, out, n, k);
}
